// MixingNetwork_63256278335521
// MI455X (gfx1250) — matrix operand order repaired and checked by reading, not compiled, not run on hardware
//
#include <hip/hip_runtime.h>

typedef __attribute__((ext_vector_type(16))) __bf16 v16bf;
typedef __attribute__((ext_vector_type(8)))  __bf16 v8bf;
typedef __attribute__((ext_vector_type(4)))  __bf16 v4bf;
typedef __attribute__((ext_vector_type(8)))  float  v8f;
typedef __attribute__((ext_vector_type(4)))  unsigned int v4u;
typedef __attribute__((ext_vector_type(8)))  int v8i;
typedef __attribute__((ext_vector_type(4)))  int v4i;

#define BN   8192
#define SDIM 512
#define HDIM 1024
#define EDIM 256
#define ADIM 32

__device__ __forceinline__ v8f vzero8() {
  v8f z;
#pragma unroll
  for (int i = 0; i < 8; ++i) z[i] = 0.0f;
  return z;
}

__device__ __forceinline__ v16bf combine16(v8bf lo, v8bf hi) {
  return __builtin_shufflevector(lo, hi, 0, 1, 2, 3, 4, 5, 6, 7, 8, 9, 10, 11, 12, 13, 14, 15);
}

// A fragment (16x32 bf16): lane m = lane&15 is row M, half = lane>>4.
// VGPR0-3 hold K = half*8 + 0..7 ; VGPR4-7 hold K = 16 + half*8 + 0..7 (ISA 7.12.2).
__device__ __forceinline__ v16bf load_fragA(const __bf16* rowk0, int half) {
  v8bf lo = *(const v8bf*)(rowk0 + half * 8);
  v8bf hi = *(const v8bf*)(rowk0 + 16 + half * 8);
  return combine16(lo, hi);
}

// B fragment (32x16 bf16) from K-contiguous source Wt[n][k]: lane n = lane&15 is col,
// per-lane K = half*8 + 0..7 then 16 + half*8 + 0..7 (same K order as A fragment).
__device__ __forceinline__ v16bf load_fragB(const __bf16* colk0, int half) {
  v8bf lo = *(const v8bf*)(colk0 + half * 8);
  v8bf hi = *(const v8bf*)(colk0 + 16 + half * 8);
  return combine16(lo, hi);
}

__device__ __forceinline__ v8f wmma_bf16(v16bf a, v16bf b, v8f c) {
  return __builtin_amdgcn_wmma_f32_16x16x32_bf16(false, a, false, b, (short)0, c, false, false);
}

// Issue one TDM DMA: 2-D tile [rows x HDIM] of bf16 from global -> LDS offset 0.
// D# layout per CDNA5 ISA 8.3/8.4 (group0: count/lds_addr/global_addr/type=2;
// group1: data_size=2B, tensor dims, tile dims, dim0 stride).
__device__ __forceinline__ void tdm_load_tile_to_lds(const __bf16* gsrc, int rows) {
  const unsigned long long ga = (unsigned long long)(const void*)gsrc;
  v4u g0;
  g0[0] = 1u;                                            // count=1, is_restore=0, gather=0
  g0[1] = 0u;                                            // lds_addr = 0 (dynamic smem base)
  g0[2] = (unsigned)(ga & 0xFFFFFFFFu);                  // global_addr[31:0]
  g0[3] = (unsigned)((ga >> 32) & 0x01FFFFFFu) | (2u << 30);  // global_addr[56:32] | type=2
  v8i g1;
  g1[0] = (1 << 16);                 // workgroup_mask=0, data_size=1 (2 bytes)
  g1[1] = (int)((unsigned)HDIM << 16);   // tensor_dim0[15:0] = 1024
  g1[2] = (int)((unsigned)BN << 16);     // tensor_dim0 hi=0 | tensor_dim1[15:0] = 8192
  g1[3] = (int)((unsigned)HDIM << 16);   // tensor_dim1 hi=0 | tile_dim0 = 1024
  g1[4] = rows;                      // tile_dim1 = rows, tile_dim2 = 0
  g1[5] = HDIM;                      // tensor_dim0_stride[31:0] = 1024
  g1[6] = 0;
  g1[7] = 0;
  v4i z4 = {0, 0, 0, 0};
#if defined(__clang_major__) && (__clang_major__ >= 23)
  v8i z8 = {0, 0, 0, 0, 0, 0, 0, 0};
  __builtin_amdgcn_tensor_load_to_lds(g0, g1, z4, z4, z8, 0);
#else
  __builtin_amdgcn_tensor_load_to_lds(g0, g1, z4, z4, 0);
#endif
}

// ---------------- fp32 -> bf16 convert (no transpose) ----------------
__global__ __launch_bounds__(256) void convert_bf16_k(const float* __restrict__ in,
                                                      __bf16* __restrict__ out, size_t n) {
  size_t i = ((size_t)blockIdx.x * 256 + threadIdx.x) * 4;
  if (i + 3 < n) {
    const float4 f = *reinterpret_cast<const float4*>(in + i);
    v4bf o;
    o[0] = (__bf16)f.x; o[1] = (__bf16)f.y; o[2] = (__bf16)f.z; o[3] = (__bf16)f.w;
    *(v4bf*)(out + i) = o;
  }
}

// ---------- fp32 [K][N] -> bf16 [N][K] transpose-convert (32x32 LDS tiles) ----------
__global__ __launch_bounds__(256) void transpose_bf16_k(const float* __restrict__ in,
                                                        __bf16* __restrict__ out,
                                                        int K, int N) {
  __shared__ float tile[32][33];
  const int x = threadIdx.x & 31;
  const int y = threadIdx.x >> 5;  // 0..7
  const int tn = blockIdx.x, tk = blockIdx.y;
#pragma unroll
  for (int yy = 0; yy < 32; yy += 8)
    tile[y + yy][x] = in[(size_t)(tk * 32 + y + yy) * N + (tn * 32 + x)];
  __syncthreads();
#pragma unroll
  for (int yy = 0; yy < 32; yy += 8)
    out[(size_t)(tn * 32 + y + yy) * K + (tk * 32 + x)] = (__bf16)tile[x][y + yy];
}

// ---------------- generic WMMA GEMM: out = act(A @ Wt^T + bias) ----------------
// A: bf16 [M][K] row-major; Wt: bf16 [N][K] (pre-transposed); one 16x16 tile per wave.
// 4 k-steps of fragments are loaded up front, then 4 chained WMMAs (loads of the next
// block issue while the accumulator chain drains in the matrix pipe).
template <int ACT /*0 none, 1 relu*/, int OUT_BF16>
__global__ __launch_bounds__(256) void gemm_bias_act_k(const __bf16* __restrict__ A,
                                                       const __bf16* __restrict__ Wt,
                                                       const float* __restrict__ bias,
                                                       void* __restrict__ out,
                                                       int M, int N, int K) {
  const int wave = threadIdx.x >> 5;
  const int lane = threadIdx.x & 31;
  const int half = lane >> 4;
  const int m = lane & 15;
  const int tilesN = N >> 4;
  const int tile = blockIdx.x * 8 + wave;
  const int tm = tile / tilesN;
  const int tn = tile - tm * tilesN;
  if (tm * 16 >= M) return;  // wave-uniform exit keeps EXEC all-ones for WMMA
  const __bf16* Arow = A + (size_t)(tm * 16 + m) * K;
  const __bf16* Bcol = Wt + (size_t)(tn * 16 + m) * K;
  v8f acc = vzero8();
  for (int k0 = 0; k0 < K; k0 += 128) {
    v16bf af[4], bfr[4];
#pragma unroll
    for (int u = 0; u < 4; ++u) {
      af[u]  = load_fragA(Arow + k0 + u * 32, half);
      bfr[u] = load_fragB(Bcol + k0 + u * 32, half);
    }
#pragma unroll
    for (int u = 0; u < 4; ++u) acc = wmma_bf16(af[u], bfr[u], acc);
  }
  const float bv = bias[tn * 16 + m];
#pragma unroll
  for (int r = 0; r < 8; ++r) {
    float x = acc[r] + bv;
    if (ACT == 1) x = fmaxf(x, 0.0f);
    const size_t row = (size_t)(tm * 16 + half * 8 + r);
    const size_t col = (size_t)(tn * 16 + m);
    if (OUT_BF16) ((__bf16*)out)[row * N + col] = (__bf16)x;
    else          ((float*)out)[row * N + col] = x;
  }
}

// ---------------- fused hypernet mix: hidden = elu(sum_a q[b,a]*|w1[b,a,:]| + b1) ----------
// One workgroup: 128 batch rows; h1 tile (128x1024 bf16 = 256KB, TDM-DMAed) + q tile in LDS.
// Each wave: 2 e-tiles x 8 b-subtiles; per agent, full-K WMMA accumulation, then
// abs + q-scale in registers (w1 is never materialized in memory).
// Inner k-step is two-phase: all 8 A fragments gathered first (16x ds_load_b128), one wait,
// then 8 independent back-to-back WMMAs so the matrix pipe overlaps the next loads.
__global__ __launch_bounds__(256, 1) void mix_hidden_k(const __bf16* __restrict__ h1B,
                                                       const __bf16* __restrict__ w2t,
                                                       const float* __restrict__ w1b2,
                                                       const float* __restrict__ agent_qs,
                                                       const float* __restrict__ b1F,
                                                       float* __restrict__ hidden) {
  extern __shared__ char smem[];
  __bf16* h1s = (__bf16*)smem;                                       // [128][HDIM] @ LDS 0
  float* qs = (float*)(smem + (size_t)128 * HDIM * sizeof(__bf16));  // [128][ADIM]
  const int tid = threadIdx.x;
  const int b0 = blockIdx.x * 128;
  const int wave = tid >> 5;
  const int lane = tid & 31;
  const int half = lane >> 4;
  const int m = lane & 15;

  // Stage h1 tile with the Tensor Data Mover (one DMA, wave 0 only), q tile manually.
  if (wave == 0) tdm_load_tile_to_lds(h1B + (size_t)b0 * HDIM, 128);
  for (int i = tid; i < 128 * ADIM; i += 256) qs[i] = agent_qs[(size_t)b0 * ADIM + i];
  if (wave == 0) __builtin_amdgcn_s_wait_tensorcnt(0);
  __syncthreads();

  const __bf16* arow[8];
#pragma unroll
  for (int bs = 0; bs < 8; ++bs) arow[bs] = h1s + (size_t)(bs * 16 + m) * HDIM;

#pragma unroll 1
  for (int et = wave * 2; et < wave * 2 + 2; ++et) {
    const int e0 = et * 16;
    v8f hid[8];
#pragma unroll
    for (int bs = 0; bs < 8; ++bs) hid[bs] = vzero8();
#pragma unroll 1
    for (int a = 0; a < ADIM; ++a) {
      v8f acc[8];
#pragma unroll
      for (int bs = 0; bs < 8; ++bs) acc[bs] = vzero8();
      const __bf16* Bcol = w2t + (size_t)(a * EDIM + e0 + m) * HDIM;
      if (a + 1 < ADIM)  // prefetch next agent's weight column (L2-resident stream)
        __builtin_prefetch(w2t + (size_t)((a + 1) * EDIM + e0 + m) * HDIM + half * 16, 0, 0);
#pragma unroll 2
      for (int k0 = 0; k0 < HDIM; k0 += 32) {
        const v16bf bf = load_fragB(Bcol + k0, half);
        v16bf af[8];
#pragma unroll
        for (int bs = 0; bs < 8; ++bs) af[bs] = load_fragA(arow[bs] + k0, half);
#pragma unroll
        for (int bs = 0; bs < 8; ++bs) acc[bs] = wmma_bf16(af[bs], bf, acc[bs]);
      }
      const float bb = w1b2[a * EDIM + e0 + m];
#pragma unroll
      for (int bs = 0; bs < 8; ++bs)
#pragma unroll
        for (int r = 0; r < 8; ++r)
          hid[bs][r] += qs[(bs * 16 + half * 8 + r) * ADIM + a] * fabsf(acc[bs][r] + bb);
    }
#pragma unroll
    for (int bs = 0; bs < 8; ++bs)
#pragma unroll
      for (int r = 0; r < 8; ++r) {
        const size_t row = (size_t)(b0 + bs * 16 + half * 8 + r);
        const float x = hid[bs][r] + b1F[row * EDIM + e0 + m];
        hidden[row * EDIM + e0 + m] = x > 0.0f ? x : (__expf(x) - 1.0f);
      }
  }
}

// ---------------- v = relu(st@v_w1+b) @ v_w2 + v_b2 (second layer only) ----------------
__global__ __launch_bounds__(256) void v_reduce_k(const float* __restrict__ vhF,
                                                  const float* __restrict__ v_w2,
                                                  const float* __restrict__ v_b2,
                                                  float* __restrict__ vF) {
  const int wave = threadIdx.x >> 5;
  const int lane = threadIdx.x & 31;
  const int row = blockIdx.x * 8 + wave;
  const float* vh = vhF + (size_t)row * EDIM;
  float s = 0.0f;
#pragma unroll
  for (int e = lane; e < EDIM; e += 32) s += vh[e] * v_w2[e];
  for (int mask = 16; mask >= 1; mask >>= 1) s += __shfl_xor(s, mask, 32);
  if (lane == 0) vF[row] = s + v_b2[0];
}

// ---------------- q_tot = hidden . |hf@hwf_w2 + b2| + v  (fused final layer) ----------------
__global__ __launch_bounds__(256) void final_qtot_k(const __bf16* __restrict__ hfB,
                                                    const __bf16* __restrict__ wft,
                                                    const float* __restrict__ wfb2,
                                                    const float* __restrict__ hidden,
                                                    const float* __restrict__ vF,
                                                    float* __restrict__ qtot) {
  const int wave = threadIdx.x >> 5;
  const int lane = threadIdx.x & 31;
  const int half = lane >> 4;
  const int m = lane & 15;
  const int b0 = (blockIdx.x * 8 + wave) * 16;
  const __bf16* Arow = hfB + (size_t)(b0 + m) * HDIM;
  float part[8];
#pragma unroll
  for (int r = 0; r < 8; ++r) part[r] = 0.0f;
#pragma unroll 1
  for (int et = 0; et < 16; ++et) {
    const int e0 = et * 16;
    const __bf16* Bcol = wft + (size_t)(e0 + m) * HDIM;
    v8f acc = vzero8();
    for (int k0 = 0; k0 < HDIM; k0 += 128) {
      v16bf af[4], bfr[4];
#pragma unroll
      for (int u = 0; u < 4; ++u) {
        af[u]  = load_fragA(Arow + k0 + u * 32, half);
        bfr[u] = load_fragB(Bcol + k0 + u * 32, half);
      }
#pragma unroll
      for (int u = 0; u < 4; ++u) acc = wmma_bf16(af[u], bfr[u], acc);
    }
    const float bv = wfb2[e0 + m];
#pragma unroll
    for (int r = 0; r < 8; ++r) {
      const size_t row = (size_t)(b0 + half * 8 + r);
      part[r] += fabsf(acc[r] + bv) * hidden[row * EDIM + e0 + m];
    }
  }
#pragma unroll
  for (int r = 0; r < 8; ++r) {
    float s = part[r];
    for (int mask = 1; mask < 16; mask <<= 1) s += __shfl_xor(s, mask, 32);
    if (m == 0) {
      const int row = b0 + half * 8 + r;
      qtot[row] = s + vF[row];
    }
  }
}

extern "C" void kernel_launch(void* const* d_in, const int* in_sizes, int n_in,
                              void* d_out, int out_size, void* d_ws, size_t ws_size,
                              hipStream_t stream) {
  (void)in_sizes; (void)n_in; (void)out_size; (void)ws_size;
  const float* agent_qs = (const float*)d_in[0];
  const float* states   = (const float*)d_in[1];
  const float* hw1_w1   = (const float*)d_in[2];
  const float* hw1_b1   = (const float*)d_in[3];
  const float* hw1_w2   = (const float*)d_in[4];
  const float* hw1_b2   = (const float*)d_in[5];
  const float* hb1_w    = (const float*)d_in[6];
  const float* hb1_b    = (const float*)d_in[7];
  const float* hwf_w1   = (const float*)d_in[8];
  const float* hwf_b1   = (const float*)d_in[9];
  const float* hwf_w2   = (const float*)d_in[10];
  const float* hwf_b2   = (const float*)d_in[11];
  const float* v_w1     = (const float*)d_in[12];
  const float* v_b1     = (const float*)d_in[13];
  const float* v_w2     = (const float*)d_in[14];
  const float* v_b2     = (const float*)d_in[15];
  float* qtot = (float*)d_out;

  // workspace carve-up (~83 MB total)
  char* p = (char*)d_ws;
  auto take = [&p](size_t bytes) { char* r = p; p += (bytes + 255) & ~(size_t)255; return r; };
  __bf16* stB    = (__bf16*)take((size_t)BN * SDIM * 2);
  __bf16* w1t    = (__bf16*)take((size_t)HDIM * SDIM * 2);
  __bf16* wf1t   = (__bf16*)take((size_t)HDIM * SDIM * 2);
  __bf16* hb1t   = (__bf16*)take((size_t)EDIM * SDIM * 2);
  __bf16* vw1t   = (__bf16*)take((size_t)EDIM * SDIM * 2);
  __bf16* w2t    = (__bf16*)take((size_t)(EDIM * ADIM) * HDIM * 2);
  __bf16* wft    = (__bf16*)take((size_t)EDIM * HDIM * 2);
  __bf16* h1B    = (__bf16*)take((size_t)BN * HDIM * 2);
  __bf16* hfB    = (__bf16*)take((size_t)BN * HDIM * 2);
  float*  b1F    = (float*)take((size_t)BN * EDIM * 4);
  float*  vhF    = (float*)take((size_t)BN * EDIM * 4);
  float*  hidden = (float*)take((size_t)BN * EDIM * 4);
  float*  vF     = (float*)take((size_t)BN * 4);

  // 1) precision conversion + weight transposition ([K][N] f32 -> [N][K] bf16)
  convert_bf16_k<<<(size_t)BN * SDIM / 4 / 256, 256, 0, stream>>>(states, stB, (size_t)BN * SDIM);
  transpose_bf16_k<<<dim3(HDIM / 32, SDIM / 32), 256, 0, stream>>>(hw1_w1, w1t, SDIM, HDIM);
  transpose_bf16_k<<<dim3(HDIM / 32, SDIM / 32), 256, 0, stream>>>(hwf_w1, wf1t, SDIM, HDIM);
  transpose_bf16_k<<<dim3(EDIM / 32, SDIM / 32), 256, 0, stream>>>(hb1_w, hb1t, SDIM, EDIM);
  transpose_bf16_k<<<dim3(EDIM / 32, SDIM / 32), 256, 0, stream>>>(v_w1, vw1t, SDIM, EDIM);
  transpose_bf16_k<<<dim3(EDIM * ADIM / 32, HDIM / 32), 256, 0, stream>>>(hw1_w2, w2t, HDIM, EDIM * ADIM);
  transpose_bf16_k<<<dim3(EDIM / 32, HDIM / 32), 256, 0, stream>>>(hwf_w2, wft, HDIM, EDIM);

  // 2) stage-1 GEMMs (WMMA bf16)
  gemm_bias_act_k<1, 1><<<(BN / 16) * (HDIM / 16) / 8, 256, 0, stream>>>(stB, w1t, hw1_b1, h1B, BN, HDIM, SDIM);
  gemm_bias_act_k<1, 1><<<(BN / 16) * (HDIM / 16) / 8, 256, 0, stream>>>(stB, wf1t, hwf_b1, hfB, BN, HDIM, SDIM);
  gemm_bias_act_k<0, 0><<<(BN / 16) * (EDIM / 16) / 8, 256, 0, stream>>>(stB, hb1t, hb1_b, b1F, BN, EDIM, SDIM);
  gemm_bias_act_k<1, 0><<<(BN / 16) * (EDIM / 16) / 8, 256, 0, stream>>>(stB, vw1t, v_b1, vhF, BN, EDIM, SDIM);

  // 3) value head second layer
  v_reduce_k<<<BN / 8, 256, 0, stream>>>(vhF, v_w2, v_b2, vF);

  // 4) fused 137-GFLOP hypernet GEMM + abs + agent mix + elu (w1 stays in registers;
  //    h1 tile staged into LDS by the Tensor Data Mover)
  const size_t mixShmem = (size_t)128 * HDIM * sizeof(__bf16) + (size_t)128 * ADIM * sizeof(float);
  mix_hidden_k<<<BN / 128, 256, mixShmem, stream>>>(h1B, w2t, hw1_b2, agent_qs, b1F, hidden);

  // 5) fused final layer: |hf@hwf_w2+b2| dot hidden + v
  final_qtot_k<<<(BN / 16) / 8, 256, 0, stream>>>(hfB, wft, hwf_b2, hidden, vF, qtot);
}
